// GLAAttention_35742717837709
// MI455X (gfx1250) — matrix operand order repaired and checked by reading, not compiled, not run on hardware
//
#include <hip/hip_runtime.h>
#include <cstdint>
#include <cstddef>

#define DEVINL __device__ __forceinline__

typedef __attribute__((ext_vector_type(16))) __bf16 v16bf;
typedef __attribute__((ext_vector_type(8)))  float  v8f;

// ---------------- problem sizes ----------------
constexpr int Bb  = 8;
constexpr int Ss  = 4096;
constexpr int Hh  = 768;
constexpr int CB  = 64;          // block size
constexpr int NBk = Ss / CB;     // 64 blocks
constexpr int MR  = Bb * Ss;     // 32768 rows
constexpr int NQKV = 3 * Hh;     // 2304 fused QKV columns
constexpr int KSTR = 776;        // padded H stride for LDS (776%8==0, (776/2)%64==4 -> conflict-free)

// ---------------- workspace layout (bytes) ----------------
constexpr size_t OFF_XBF = 0;                                   // x bf16        [32768][768]
constexpr size_t OFF_WT  = OFF_XBF + (size_t)MR * Hh * 2;       // Wqkv^T bf16   [2304][768]
constexpr size_t OFF_QB  = OFF_WT  + (size_t)NQKV * Hh * 2;     // Q bf16        [B][S][H]
constexpr size_t OFF_KT  = OFF_QB  + (size_t)MR * Hh * 2;       // K^T bf16      [B][NB][H][64]
constexpr size_t OFF_VT  = OFF_KT  + (size_t)Bb * NBk * Hh * CB * 2; // V^T bf16 [B][NB][H][64]
constexpr size_t OFF_WS  = OFF_VT  + (size_t)Bb * NBk * Hh * CB * 2; // softmax w [B][NB][64][64] bf16
constexpr size_t OFF_H16 = OFF_WS  + (size_t)Bb * NBk * CB * CB * 2; // x@W1 f32  [32768][16]
constexpr size_t OFF_AG  = OFF_H16 + (size_t)MR * 16 * 4;       // gate a f32    [B][H]

// ---------------- helpers ----------------
struct Frag32 { uint4 lo, hi; };

static DEVINL v16bf mkfrag(uint4 lo, uint4 hi) {
  Frag32 t; t.lo = lo; t.hi = hi;
  return __builtin_bit_cast(v16bf, t);
}

static DEVINL unsigned short f2bf(float f) {          // RNE f32 -> bf16
  unsigned int u = __float_as_uint(f);
  u += 0x7fffu + ((u >> 16) & 1u);
  return (unsigned short)(u >> 16);
}
static DEVINL unsigned int pk2(float a, float b) {
  return (unsigned int)f2bf(a) | ((unsigned int)f2bf(b) << 16);
}

static DEVINL v8f wmma_bf16(v16bf a, v16bf b, v8f c) {
  // D = A(16x32 bf16) * B(32x16 bf16) + C(16x16 f32)
  return __builtin_amdgcn_wmma_f32_16x16x32_bf16(false, a, false, b, (short)0, c, false, false);
}

// ---------------- kernel X: x f32 -> bf16 ----------------
__global__ __launch_bounds__(256) void k_cvt_x(const float* __restrict__ x,
                                               unsigned short* __restrict__ xbf) {
  size_t i = ((size_t)blockIdx.x * 256 + threadIdx.x) * 4;
  float4 f = *reinterpret_cast<const float4*>(x + i);
  uint2 p; p.x = pk2(f.x, f.y); p.y = pk2(f.z, f.w);
  *reinterpret_cast<uint2*>(xbf + i) = p;
}

// ---------------- kernel W: [Wq|Wk|Wv] -> transposed bf16 Wt[n][k] ----------------
__global__ __launch_bounds__(256) void k_cvt_w2(const float* __restrict__ Wq,
                                                const float* __restrict__ Wk,
                                                const float* __restrict__ Wv,
                                                unsigned short* __restrict__ Wt) {
  int idx = blockIdx.x * 256 + threadIdx.x;   // 0 .. 2304*768-1
  int n = idx / Hh;
  int k = idx - n * Hh;
  int sec = n / Hh;        // 0:Q 1:K 2:V
  int hn  = n - sec * Hh;
  const float* Wsrc = (sec == 0) ? Wq : (sec == 1) ? Wk : Wv;
  Wt[idx] = f2bf(Wsrc[(size_t)k * Hh + hn]);  // Wt[n][k] = W[k][hn]
}

// ---------------- kernel A: fused QKV GEMM (bf16 WMMA) ----------------
// grid (36 = 2304/64, 512 = 32768/64), 128 threads (4 waves), 64x64 tile / WG
__global__ __launch_bounds__(128) void k_qkv(const unsigned short* __restrict__ xbf,
                                             const unsigned short* __restrict__ Wt,
                                             unsigned short* __restrict__ Qb,
                                             unsigned short* __restrict__ Kt,
                                             unsigned short* __restrict__ Vt) {
  const int bx = blockIdx.x, by = blockIdx.y;
  const int tid = threadIdx.x;
  const int w = tid >> 5, lane = tid & 31, l16 = lane & 15, hi = lane >> 4;

  v8f acc[4] = {};
  const uint4* a4 = reinterpret_cast<const uint4*>(xbf + (size_t)(by * 64 + w * 16 + l16) * Hh);
  const uint4* b4[4];
#pragma unroll
  for (int dt = 0; dt < 4; ++dt)
    b4[dt] = reinterpret_cast<const uint4*>(Wt + (size_t)(bx * 64 + dt * 16 + l16) * Hh);

  for (int kk = 0; kk < Hh / 32; ++kk) {
    v16bf af = mkfrag(a4[kk * 4 + hi], a4[kk * 4 + 2 + hi]);
#pragma unroll
    for (int dt = 0; dt < 4; ++dt) {
      v16bf bf = mkfrag(b4[dt][kk * 4 + hi], b4[dt][kk * 4 + 2 + hi]);
      acc[dt] = wmma_bf16(af, bf, acc[dt]);
    }
  }

  const int s0  = by * 64 + w * 16 + hi * 8;       // first output row of this lane
  const int sec = bx / 12;                         // 12 column-tiles per 768-wide section
  const int b_  = by >> 6, nb = by & 63;           // batch / block of this 64-row strip
#pragma unroll
  for (int dt = 0; dt < 4; ++dt) {
    int hcol = (bx % 12) * 64 + dt * 16 + l16;
    if (sec == 0) {                                // Q natural [B*S][H]
#pragma unroll
      for (int v = 0; v < 8; ++v)
        Qb[(size_t)(s0 + v) * Hh + hcol] = f2bf(acc[dt][v]);
    } else {                                       // K/V -> block-transposed [b][nb][h][c]
      unsigned short* dst = (sec == 1) ? Kt : Vt;
      size_t base = ((size_t)((b_ * NBk + nb) * Hh + hcol)) * CB + (w * 16 + hi * 8);
      uint4 p;
      p.x = pk2(acc[dt][0], acc[dt][1]);
      p.y = pk2(acc[dt][2], acc[dt][3]);
      p.z = pk2(acc[dt][4], acc[dt][5]);
      p.w = pk2(acc[dt][6], acc[dt][7]);
      *reinterpret_cast<uint4*>(dst + base) = p;
    }
  }
}

// ---------------- kernel G1: h16 = x @ W1  (f32) ----------------
__global__ __launch_bounds__(256) void k_g1(const float* __restrict__ x,
                                            const float* __restrict__ W1,
                                            float* __restrict__ h16) {
  __shared__ float w1s[Hh * 16];
  for (int i = threadIdx.x; i < Hh * 16; i += 256) w1s[i] = W1[i];
  __syncthreads();
  int s = blockIdx.x * 256 + threadIdx.x;
  const float* xr = x + (size_t)s * Hh;
  float acc[16] = {};
  for (int j = 0; j < Hh; ++j) {
    float xv = xr[j];
#pragma unroll
    for (int t = 0; t < 16; ++t) acc[t] = fmaf(xv, w1s[j * 16 + t], acc[t]);
  }
#pragma unroll
  for (int t = 0; t < 16; ++t) h16[(size_t)s * 16 + t] = acc[t];
}

// ---------------- kernel G2: a[b,h] = mean_s sigmoid(h16@W2 + b)^(1/16) ----------------
// one wave per (b,h); 8 waves per WG
__global__ __launch_bounds__(256) void k_g2(const float* __restrict__ h16,
                                            const float* __restrict__ W2,
                                            const float* __restrict__ bias,
                                            float* __restrict__ aG) {
  int gw = blockIdx.x * 8 + (threadIdx.x >> 5);
  int lane = threadIdx.x & 31;
  int b_ = gw / Hh, h = gw - b_ * Hh;
  float w2c[16];
#pragma unroll
  for (int j = 0; j < 16; ++j) w2c[j] = W2[(size_t)j * Hh + h];
  float bh = bias[h];
  float acc = 0.f;
  for (int s = lane; s < Ss; s += 32) {
    const float* hr = h16 + ((size_t)b_ * Ss + s) * 16;
    float d = bh;
#pragma unroll
    for (int j = 0; j < 16; ++j) d = fmaf(hr[j], w2c[j], d);
    float sig = 1.f / (1.f + __expf(-d));
    acc += sqrtf(sqrtf(sqrtf(sqrtf(sig))));     // sig^(1/16)
  }
#pragma unroll
  for (int off = 16; off > 0; off >>= 1) acc += __shfl_down(acc, off);
  if (lane == 0) aG[gw] = acc * (1.f / (float)Ss);
}

// ---------------- kernel B: w = softmax(q k^T * mask * scale) per (b, nb) ----------------
// grid (NB, B), 128 threads (4 waves); wave w owns row-tile mt=w, all 4 col-tiles
__global__ __launch_bounds__(128) void k_scores(const unsigned short* __restrict__ Qb,
                                                const unsigned short* __restrict__ Kt,
                                                unsigned short* __restrict__ wS) {
  const int nb = blockIdx.x, b_ = blockIdx.y;
  const int tid = threadIdx.x;
  const int w = tid >> 5, lane = tid & 31, l16 = lane & 15, hi = lane >> 4;

  __shared__ unsigned short kbn[CB * KSTR];       // kb natural [c][h] bf16 (~97KB)
  const uint4* kt4 = reinterpret_cast<const uint4*>(Kt + (size_t)(b_ * NBk + nb) * Hh * CB);
  for (int j = tid; j < Hh * 8; j += 128) {       // 8 uint4 chunks per h-row
    int h = j >> 3, cq = j & 7;
    union { uint4 q; unsigned short u[8]; } d;
    d.q = kt4[j];
#pragma unroll
    for (int e = 0; e < 8; ++e) kbn[(cq * 8 + e) * KSTR + h] = d.u[e];
  }
  __syncthreads();

  const uint4* a4 = reinterpret_cast<const uint4*>(
      Qb + (size_t)(b_ * Ss + nb * CB + w * 16 + l16) * Hh);
  v8f sc[4] = {};
  for (int kk = 0; kk < Hh / 32; ++kk) {
    v16bf af = mkfrag(a4[kk * 4 + 2 * hi], a4[kk * 4 + 2 * hi + 1]);
#pragma unroll
    for (int nt = 0; nt < 4; ++nt) {
      const unsigned short* br = kbn + (nt * 16 + l16) * KSTR + kk * 32 + hi * 16;
      v16bf bf = mkfrag(*reinterpret_cast<const uint4*>(br),
                        *reinterpret_cast<const uint4*>(br + 8));
      sc[nt] = wmma_bf16(af, bf, sc[nt]);
    }
  }

  const float scale = 0.03608439182435161f;       // 1/sqrt(768)
  unsigned short* wBlk = wS + (size_t)(b_ * NBk + nb) * CB * CB;
#pragma unroll
  for (int v = 0; v < 8; ++v) {
    int c = w * 16 + hi * 8 + v;                  // score row
    float vals[4];
#pragma unroll
    for (int nt = 0; nt < 4; ++nt) {
      int cp = nt * 16 + l16;                     // score col
      vals[nt] = (cp <= c) ? sc[nt][v] * scale : 0.f;   // multiplicative mask
    }
    float m = fmaxf(fmaxf(vals[0], vals[1]), fmaxf(vals[2], vals[3]));
#pragma unroll
    for (int off = 1; off < 16; off <<= 1) m = fmaxf(m, __shfl_xor(m, off));
    float p[4], sum = 0.f;
#pragma unroll
    for (int nt = 0; nt < 4; ++nt) { p[nt] = __expf(vals[nt] - m); sum += p[nt]; }
#pragma unroll
    for (int off = 1; off < 16; off <<= 1) sum += __shfl_xor(sum, off);
    float inv = 1.f / sum;
    unsigned short* wrow = wBlk + (size_t)c * CB;
#pragma unroll
    for (int nt = 0; nt < 4; ++nt) wrow[nt * 16 + l16] = f2bf(p[nt] * inv);
  }
}

// ---------------- kernel C: recurrent scan ----------------
// grid (12 = H/64 d-slices, B), 256 threads (8 waves).
// State slice St[768][64] f32 lives in WMMA accumulators: wave w owns h-tiles {w, w+8,...,w+40} x 4 d-tiles.
// bf16 mirror St^T[d][h] in LDS feeds q@St B-fragments.
__global__ __launch_bounds__(256) void k_scan(const unsigned short* __restrict__ Qb,
                                              const unsigned short* __restrict__ Kt,
                                              const unsigned short* __restrict__ Vt,
                                              const unsigned short* __restrict__ wS,
                                              const float* __restrict__ aG,
                                              float* __restrict__ out) {
  const int dsl = blockIdx.x * 64;                // d-slice base
  const int b_  = blockIdx.y;
  const int tid = threadIdx.x;
  const int w = tid >> 5, lane = tid & 31, l16 = lane & 15, hi = lane >> 4;

  __shared__ unsigned short StT[CB * KSTR];       // St^T[d][h] bf16 mirror (~97KB)

  float av[4];
#pragma unroll
  for (int dt = 0; dt < 4; ++dt) av[dt] = aG[b_ * Hh + dsl + dt * 16 + l16];

  v8f st[6][4] = {};                              // 192 VGPRs of f32 state / lane

  for (int nb = 0; nb < NBk; ++nb) {
    const unsigned short* KtB = Kt + (size_t)(b_ * NBk + nb) * Hh * CB;
    const unsigned short* VtB = Vt + (size_t)(b_ * NBk + nb) * Hh * CB;

    // B-fragments of v^T for this d-slice (reused by update and by w@v)
    v16bf vb0[4], vb1[4];
#pragma unroll
    for (int dt = 0; dt < 4; ++dt) {
      const uint4* vt = reinterpret_cast<const uint4*>(VtB + (size_t)(dsl + dt * 16 + l16) * CB);
      vb0[dt] = mkfrag(vt[hi], vt[2 + hi]);
      vb1[dt] = mkfrag(vt[4 + hi], vt[6 + hi]);
    }

    // ---- state update: St = St * a + k^T v ----
#pragma unroll
    for (int i = 0; i < 6; ++i) {
      int ht = w + i * 8;
      const uint4* kt = reinterpret_cast<const uint4*>(KtB + (size_t)(ht * 16 + l16) * CB);
      v16bf ka0 = mkfrag(kt[hi], kt[2 + hi]);
      v16bf ka1 = mkfrag(kt[4 + hi], kt[6 + hi]);
#pragma unroll
      for (int dt = 0; dt < 4; ++dt) {
        v8f acc = st[i][dt] * av[dt];             // per-lane decay a[d]
        acc = wmma_bf16(ka0, vb0[dt], acc);
        acc = wmma_bf16(ka1, vb1[dt], acc);
        st[i][dt] = acc;
      }
    }

    __syncthreads();                              // prior out-phase readers done
    // ---- write bf16 mirror St^T[d][h] (packed b128 LDS stores) ----
#pragma unroll
    for (int i = 0; i < 6; ++i) {
      int ht = w + i * 8;
#pragma unroll
      for (int dt = 0; dt < 4; ++dt) {
        uint4 p;
        p.x = pk2(st[i][dt][0], st[i][dt][1]);
        p.y = pk2(st[i][dt][2], st[i][dt][3]);
        p.z = pk2(st[i][dt][4], st[i][dt][5]);
        p.w = pk2(st[i][dt][6], st[i][dt][7]);
        *reinterpret_cast<uint4*>(&StT[(dt * 16 + l16) * KSTR + ht * 16 + hi * 8]) = p;
      }
    }
    __syncthreads();

    // ---- out = q @ St + w @ v  (2 output tiles per wave) ----
    const unsigned short* wB = wS + (size_t)(b_ * NBk + nb) * CB * CB;
#pragma unroll
    for (int j = 0; j < 2; ++j) {
      int t = w * 2 + j, ct = t >> 2, dt = t & 3;
      v8f acc = {};
      const uint4* wr = reinterpret_cast<const uint4*>(wB + (size_t)(ct * 16 + l16) * CB);
      acc = wmma_bf16(mkfrag(wr[hi], wr[2 + hi]), vb0[dt], acc);
      acc = wmma_bf16(mkfrag(wr[4 + hi], wr[6 + hi]), vb1[dt], acc);

      const uint4* qr = reinterpret_cast<const uint4*>(
          Qb + (size_t)(b_ * Ss + nb * CB + ct * 16 + l16) * Hh);
      const unsigned short* srow = &StT[(dt * 16 + l16) * KSTR];
      for (int kk = 0; kk < Hh / 32; ++kk) {
        v16bf af = mkfrag(qr[kk * 4 + 2 * hi], qr[kk * 4 + 2 * hi + 1]);
        v16bf bf = mkfrag(*reinterpret_cast<const uint4*>(srow + kk * 32 + hi * 16),
                          *reinterpret_cast<const uint4*>(srow + kk * 32 + hi * 16 + 8));
        acc = wmma_bf16(af, bf, acc);
      }
      float* orow = out + ((size_t)(b_ * Ss + nb * CB + ct * 16 + hi * 8)) * Hh
                    + dsl + dt * 16 + l16;
#pragma unroll
      for (int v = 0; v < 8; ++v) orow[(size_t)v * Hh] = acc[v];
    }
  }
}

// ---------------- launcher ----------------
extern "C" void kernel_launch(void* const* d_in, const int* in_sizes, int n_in,
                              void* d_out, int out_size, void* d_ws, size_t ws_size,
                              hipStream_t stream) {
  (void)in_sizes; (void)n_in; (void)out_size; (void)ws_size;
  const float* x  = (const float*)d_in[0];
  const float* Wq = (const float*)d_in[1];
  const float* Wk = (const float*)d_in[2];
  const float* Wv = (const float*)d_in[3];
  const float* W1 = (const float*)d_in[4];
  const float* W2 = (const float*)d_in[5];
  const float* bs = (const float*)d_in[6];

  char* ws = (char*)d_ws;
  unsigned short* xbf = (unsigned short*)(ws + OFF_XBF);
  unsigned short* Wt  = (unsigned short*)(ws + OFF_WT);
  unsigned short* Qb  = (unsigned short*)(ws + OFF_QB);
  unsigned short* Kt  = (unsigned short*)(ws + OFF_KT);
  unsigned short* Vt  = (unsigned short*)(ws + OFF_VT);
  unsigned short* wSc = (unsigned short*)(ws + OFF_WS);
  float*          h16 = (float*)(ws + OFF_H16);
  float*          aG  = (float*)(ws + OFF_AG);

  k_cvt_x <<< (MR * Hh) / 1024, 256, 0, stream >>> (x, xbf);
  k_cvt_w2<<< (NQKV * Hh) / 256, 256, 0, stream >>> (Wq, Wk, Wv, Wt);
  k_qkv   <<< dim3(NQKV / 64, MR / 64), 128, 0, stream >>> (xbf, Wt, Qb, Kt, Vt);
  k_g1    <<< MR / 256, 256, 0, stream >>> (x, W1, h16);
  k_g2    <<< (Bb * Hh) / 8, 256, 0, stream >>> (h16, W2, bs, aG);
  k_scores<<< dim3(NBk, Bb), 128, 0, stream >>> (Qb, Kt, wSc);
  k_scan  <<< dim3(Hh / 64, Bb), 256, 0, stream >>> (Qb, Kt, Vt, wSc, aG, (float*)d_out);
}
